// AttentionBlock_68582037782653
// MI455X (gfx1250) — matrix operand order repaired and checked by reading, not compiled, not run on hardware
//
#include <hip/hip_runtime.h>

typedef _Float16 h16;
typedef __attribute__((ext_vector_type(16))) _Float16 v16h;
typedef __attribute__((ext_vector_type(8)))  float    v8f;

// 16-byte vector type matching the async-copy builtin prototype (V4i)
typedef int b128_t __attribute__((vector_size(16)));
typedef __attribute__((address_space(1))) b128_t gb128_t;   // global (AS1)
typedef __attribute__((address_space(3))) b128_t lb128_t;   // LDS    (AS3)

#define B_   4
#define T_   2048
#define D_   2048
#define H_   16
#define HD_  128
#define BT_  (B_ * T_)
#define DFF_ (4 * D_)

#if __has_builtin(__builtin_amdgcn_global_load_async_to_lds_b128)
#define ASYNC_COPY 1
#else
#define ASYNC_COPY 0
#endif

// ---------------------------------------------------------------------------
// async global(16B)->LDS copy; falls back to sync load/store if unavailable
// ---------------------------------------------------------------------------
__device__ __forceinline__ void copy_b128(h16* lds, const h16* g) {
#if ASYNC_COPY
  __builtin_amdgcn_global_load_async_to_lds_b128(
      (gb128_t*)(size_t)g, (lb128_t*)lds, 0, 0);
#else
  *(float4*)lds = *(const float4*)g;
#endif
}

__device__ __forceinline__ void wait_async() {
#if ASYNC_COPY
#if __has_builtin(__builtin_amdgcn_s_wait_asynccnt)
  __builtin_amdgcn_s_wait_asynccnt(0);
#else
  asm volatile("s_wait_asynccnt 0" ::: "memory");
#endif
#endif
}

// stage a 128-row x 32-half tile (LDS pitch 40 halfs), 256 threads
__device__ __forceinline__ void stage128x32(h16* dst, const h16* src, int ld) {
  const int tid = threadIdx.x;
#pragma unroll
  for (int rep = 0; rep < 2; ++rep) {
    const int it = tid + rep * 256;
    const int i  = it >> 2;
    const int jc = (it & 3) * 8;
    copy_b128(dst + i * 40 + jc, src + (size_t)i * ld + jc);
  }
}
// stage a 32-row x 128-half tile (LDS pitch 136 halfs), 256 threads
__device__ __forceinline__ void stage32x128(h16* dst, const h16* src, int ld) {
  const int tid = threadIdx.x;
#pragma unroll
  for (int rep = 0; rep < 2; ++rep) {
    const int it = tid + rep * 256;
    const int i  = it >> 4;
    const int jc = (it & 15) * 8;
    copy_b128(dst + i * 136 + jc, src + (size_t)i * ld + jc);
  }
}

// ---------------------------------------------------------------------------
// WMMA fragment helpers (CDNA5 ISA §7.12.2, wave32)
// ---------------------------------------------------------------------------
__device__ __forceinline__ v16h load_frag_a(const h16* p, int ld, int row0) {
  const int lane = threadIdx.x & 31;
  const int sub  = lane & 15;
  const int hs   = lane >> 4;
  const h16* r = p + (size_t)(row0 + sub) * ld + hs * 16;
  v16h f;
#pragma unroll
  for (int g = 0; g < 8; ++g) {
    const int k = 2 * g;
    f[2 * g]     = r[k];
    f[2 * g + 1] = r[k + 1];
  }
  return f;
}

__device__ __forceinline__ v16h load_frag_b(const h16* p, int ld, int row0) {
  const int lane = threadIdx.x & 31;
  const int sub  = lane & 15;
  const int hs   = lane >> 4;
  const h16* r = p + (size_t)(row0 + sub) * ld + hs * 16;
  v16h f;
#pragma unroll
  for (int g = 0; g < 8; ++g) {
    f[2 * g]     = r[2 * g];
    f[2 * g + 1] = r[2 * g + 1];
  }
  return f;
}

__device__ __forceinline__ v8f wmma_f16(v16h a, v16h b, v8f c) {
  return __builtin_amdgcn_wmma_f32_16x16x32_f16(false, a, false, b, (short)0, c,
                                                false, false);
}

// ---------------------------------------------------------------------------
// Tiled transpose + f32->f16 cast: dst[n*K + k] = (h16) src[k*N + n]
// 32x32 tiles, 256 threads; blockIdx.z = batch (head) of size K*N
// ---------------------------------------------------------------------------
__global__ void k_cast_t(const float* __restrict__ src, h16* __restrict__ dst,
                         int K, int N) {
  __shared__ float tile[32][33];
  const int k0 = blockIdx.y * 32;
  const int n0 = blockIdx.x * 32;
  const size_t zo = (size_t)blockIdx.z * K * N;
  const int tx = threadIdx.x & 31;
  const int ty = threadIdx.x >> 5;
#pragma unroll
  for (int r = 0; r < 4; ++r)
    tile[ty + r * 8][tx] = src[zo + (size_t)(k0 + ty + r * 8) * N + n0 + tx];
  __syncthreads();
#pragma unroll
  for (int r = 0; r < 4; ++r)
    dst[zo + (size_t)(n0 + ty + r * 8) * K + k0 + tx] = (h16)tile[tx][ty + r * 8];
}

// ---------------------------------------------------------------------------
// LayerNorm (optionally fused residual add)
// ---------------------------------------------------------------------------
template <int ADD_RES>
__global__ void k_ln(const float* __restrict__ x, const float* __restrict__ res,
                     const float* __restrict__ g, const float* __restrict__ be,
                     h16* __restrict__ out_h, float* __restrict__ out_res) {
  __shared__ float red[8];
  const int row  = blockIdx.x;
  const int tid  = threadIdx.x;
  const int lane = tid & 31;
  const int wave = tid >> 5;
  const float* xr = x + (size_t)row * D_;

  float v[8];
  float s = 0.f;
#pragma unroll
  for (int j = 0; j < 8; ++j) {
    const int c = tid + j * 256;
    float t = xr[c];
    if (ADD_RES) t += res[(size_t)row * D_ + c];
    v[j] = t;
    s += t;
  }
#pragma unroll
  for (int m = 16; m >= 1; m >>= 1) s += __shfl_xor(s, m, 32);
  if (lane == 0) red[wave] = s;
  __syncthreads();
  float tot = 0.f;
#pragma unroll
  for (int w = 0; w < 8; ++w) tot += red[w];
  const float mean = tot * (1.0f / D_);
  __syncthreads();

  float vs = 0.f;
#pragma unroll
  for (int j = 0; j < 8; ++j) {
    const float d = v[j] - mean;
    vs += d * d;
  }
#pragma unroll
  for (int m = 16; m >= 1; m >>= 1) vs += __shfl_xor(vs, m, 32);
  if (lane == 0) red[wave] = vs;
  __syncthreads();
  tot = 0.f;
#pragma unroll
  for (int w = 0; w < 8; ++w) tot += red[w];
  const float inv = rsqrtf(tot * (1.0f / D_) + 1e-5f);

#pragma unroll
  for (int j = 0; j < 8; ++j) {
    const int c = tid + j * 256;
    const float y = (v[j] - mean) * inv * g[c] + be[c];
    out_h[(size_t)row * D_ + c] = (h16)y;
    if (ADD_RES) out_res[(size_t)row * D_ + c] = v[j];
  }
}

// ---------------------------------------------------------------------------
// f16 GEMM, B pre-transposed: C[M,N] = A[M,K] * Bt[N,K]^T (+bias)(+res)(ReLU)
// 128x128x32 tiles, 8 waves, wave tile 32x64; double-buffered async staging
// ---------------------------------------------------------------------------
template <int OUT_F16, int RELU, int ADD_RES>
__global__ __launch_bounds__(256, 1)
void k_gemm(const h16* __restrict__ A, const h16* __restrict__ Bt,
            const float* __restrict__ bias, const float* __restrict__ res,
            void* __restrict__ Cout, int M, int N, int K) {
  __shared__ h16 As[2][128][40];
  __shared__ h16 Bs[2][128][40];
  const int tid  = threadIdx.x;
  const int m0   = blockIdx.y * 128;
  const int n0   = blockIdx.x * 128;
  const int wave = tid >> 5;
  const int wm   = wave & 3;
  const int wn   = wave >> 2;
  const int lane = tid & 31;
  const int sub  = lane & 15;
  const int hs   = lane >> 4;

  v8f acc[2][4];
#pragma unroll
  for (int mi = 0; mi < 2; ++mi)
#pragma unroll
    for (int ni = 0; ni < 4; ++ni)
#pragma unroll
      for (int r = 0; r < 8; ++r) acc[mi][ni][r] = 0.f;

  stage128x32(&As[0][0][0], A + (size_t)m0 * K, K);
  stage128x32(&Bs[0][0][0], Bt + (size_t)n0 * K, K);
  wait_async();
  __syncthreads();

  int buf = 0;
  for (int k0 = 0; k0 < K; k0 += 32) {
    if (k0 + 32 < K) {  // prefetch next tile into the other buffer
      stage128x32(&As[buf ^ 1][0][0], A + (size_t)m0 * K + k0 + 32, K);
      stage128x32(&Bs[buf ^ 1][0][0], Bt + (size_t)n0 * K + k0 + 32, K);
    }
    v16h af[2], bf[4];
#pragma unroll
    for (int mi = 0; mi < 2; ++mi)
      af[mi] = load_frag_a(&As[buf][0][0], 40, wm * 32 + mi * 16);
#pragma unroll
    for (int ni = 0; ni < 4; ++ni)
      bf[ni] = load_frag_b(&Bs[buf][0][0], 40, wn * 64 + ni * 16);
#pragma unroll
    for (int mi = 0; mi < 2; ++mi)
#pragma unroll
      for (int ni = 0; ni < 4; ++ni)
        acc[mi][ni] = wmma_f16(af[mi], bf[ni], acc[mi][ni]);
    wait_async();
    __syncthreads();
    buf ^= 1;
  }

#pragma unroll
  for (int mi = 0; mi < 2; ++mi)
#pragma unroll
    for (int ni = 0; ni < 4; ++ni)
#pragma unroll
      for (int r = 0; r < 8; ++r) {
        const int row = m0 + wm * 32 + mi * 16 + r + 8 * hs;
        const int col = n0 + wn * 64 + ni * 16 + sub;
        float val = acc[mi][ni][r];
        if (bias) val += bias[col];
        if (ADD_RES) val += res[(size_t)row * N + col];
        if (RELU) val = fmaxf(val, 0.f);
        if (OUT_F16) ((h16*)Cout)[(size_t)row * N + col] = (h16)val;
        else ((float*)Cout)[(size_t)row * N + col] = val;
      }
}

// ---------------------------------------------------------------------------
// QKV projection with pre-transposed weights Wt[H, HD, D].
// TRANS_OUT=0: Out[b,h,t,e] (Q,K).  TRANS_OUT=1: Out[b,h,e,t] (V).
// ---------------------------------------------------------------------------
template <int TRANS_OUT>
__global__ __launch_bounds__(256, 1)
void k_gemm_qkv(const h16* __restrict__ A, const h16* __restrict__ Wt,
                const float* __restrict__ bias, h16* __restrict__ Out) {
  __shared__ h16 As[2][128][40];
  __shared__ h16 Bs[2][128][40];
  const int tid  = threadIdx.x;
  const int head = blockIdx.y;
  const int m0   = blockIdx.x * 128;
  const int wave = tid >> 5;
  const int wm   = wave & 3;
  const int wn   = wave >> 2;
  const int lane = tid & 31;
  const int sub  = lane & 15;
  const int hs   = lane >> 4;
  const h16* Bm = Wt + (size_t)head * HD_ * D_;  // [HD, D]

  v8f acc[2][4];
#pragma unroll
  for (int mi = 0; mi < 2; ++mi)
#pragma unroll
    for (int ni = 0; ni < 4; ++ni)
#pragma unroll
      for (int r = 0; r < 8; ++r) acc[mi][ni][r] = 0.f;

  stage128x32(&As[0][0][0], A + (size_t)m0 * D_, D_);
  stage128x32(&Bs[0][0][0], Bm, D_);
  wait_async();
  __syncthreads();

  int buf = 0;
  for (int k0 = 0; k0 < D_; k0 += 32) {
    if (k0 + 32 < D_) {
      stage128x32(&As[buf ^ 1][0][0], A + (size_t)m0 * D_ + k0 + 32, D_);
      stage128x32(&Bs[buf ^ 1][0][0], Bm + k0 + 32, D_);
    }
    v16h af[2], bf[4];
#pragma unroll
    for (int mi = 0; mi < 2; ++mi)
      af[mi] = load_frag_a(&As[buf][0][0], 40, wm * 32 + mi * 16);
#pragma unroll
    for (int ni = 0; ni < 4; ++ni)
      bf[ni] = load_frag_b(&Bs[buf][0][0], 40, wn * 64 + ni * 16);
#pragma unroll
    for (int mi = 0; mi < 2; ++mi)
#pragma unroll
      for (int ni = 0; ni < 4; ++ni)
        acc[mi][ni] = wmma_f16(af[mi], bf[ni], acc[mi][ni]);
    wait_async();
    __syncthreads();
    buf ^= 1;
  }

#pragma unroll
  for (int mi = 0; mi < 2; ++mi)
#pragma unroll
    for (int ni = 0; ni < 4; ++ni)
#pragma unroll
      for (int r = 0; r < 8; ++r) {
        const int row = m0 + wm * 32 + mi * 16 + r + 8 * hs;  // global token
        const int col = wn * 64 + ni * 16 + sub;              // 0..127 = HD
        const int b = row / T_;
        const int t = row % T_;
        const float val = acc[mi][ni][r] + bias[head * HD_ + col];
        if (TRANS_OUT)
          Out[(((size_t)b * H_ + head) * HD_ + col) * T_ + t] = (h16)val;
        else
          Out[(((size_t)b * H_ + head) * T_ + t) * HD_ + col] = (h16)val;
      }
}

// ---------------------------------------------------------------------------
// Causal flash attention. V arrives transposed [B,H,HD,T] so both K and V
// stage as contiguous async b128 rows. Double-buffered over 32-key blocks.
// ---------------------------------------------------------------------------
__global__ __launch_bounds__(256, 1)
void k_attn(const h16* __restrict__ Q, const h16* __restrict__ K,
            const h16* __restrict__ Vt_g, float* __restrict__ Out) {
  __shared__ h16 Ks[2][32][136];    // [s][e] == B^T layout for S = Q K^T
  __shared__ h16 Vt[2][128][40];    // [e][s] == B^T layout for O = P V
  __shared__ h16 Ps[8][16][40];     // per-wave P tile (16 rows x 32 keys)

  const int b    = blockIdx.z;
  const int head = blockIdx.y;
  const int t0   = blockIdx.x * 128;
  const int tid  = threadIdx.x;
  const int wave = tid >> 5;
  const int lane = tid & 31;
  const int sub  = lane & 15;
  const int hs   = lane >> 4;

  const size_t base = ((size_t)b * H_ + head) * T_ * HD_;
  const h16* Qh  = Q + base;
  const h16* Kh  = K + base;
  const h16* VhT = Vt_g + base;     // [HD, T]
  const int wrow0 = t0 + wave * 16;

  v16h qf[4];
#pragma unroll
  for (int kk = 0; kk < 4; ++kk)
    qf[kk] = load_frag_a(Qh + (size_t)wrow0 * HD_ + kk * 32, HD_, 0);

  float m_run[8], l_run[8];
  v8f o[8];
#pragma unroll
  for (int r = 0; r < 8; ++r) { m_run[r] = -__builtin_inff(); l_run[r] = 0.f; }
#pragma unroll
  for (int nt = 0; nt < 8; ++nt)
#pragma unroll
    for (int r = 0; r < 8; ++r) o[nt][r] = 0.f;

  const float scale = 0.08838834764831845f;  // 1/sqrt(128)
  const int kmax = t0 + 128;                 // causal bound (exclusive)

  stage32x128(&Ks[0][0][0], Kh, HD_);        // K rows [s][e]
  stage128x32(&Vt[0][0][0], VhT, T_);        // V^T rows [e][s]
  wait_async();
  __syncthreads();

  int buf = 0;
  for (int kb = 0; kb < kmax; kb += 32) {
    if (kb + 32 < kmax) {
      stage32x128(&Ks[buf ^ 1][0][0], Kh + (size_t)(kb + 32) * HD_, HD_);
      stage128x32(&Vt[buf ^ 1][0][0], VhT + kb + 32, T_);
    }

    // S = Q K^T : 2 key sub-tiles of 16, K-dim 128 -> 4 wmma each
    v8f sacc[2];
#pragma unroll
    for (int nt = 0; nt < 2; ++nt) {
#pragma unroll
      for (int r = 0; r < 8; ++r) sacc[nt][r] = 0.f;
#pragma unroll
      for (int kk = 0; kk < 4; ++kk) {
        v16h bf = load_frag_b(&Ks[buf][0][0] + kk * 32, 136, nt * 16);
        sacc[nt] = wmma_f16(qf[kk], bf, sacc[nt]);
      }
    }

    // per-row: scale, causal mask, row-max, online-softmax update, exp, l
    // (single fused loop keeps temporaries short-lived -> no spills)
#pragma unroll
    for (int r = 0; r < 8; ++r) {
      const int row = t0 + wave * 16 + r + 8 * hs;
      float s0 = sacc[0][r] * scale;
      float s1 = sacc[1][r] * scale;
      if (kb + sub > row)      s0 = -__builtin_inff();
      if (kb + 16 + sub > row) s1 = -__builtin_inff();
      float m = fmaxf(s0, s1);
#pragma unroll
      for (int msk = 8; msk >= 1; msk >>= 1) m = fmaxf(m, __shfl_xor(m, msk, 32));
      const float mn = fmaxf(m_run[r], m);
      const float al = __expf(m_run[r] - mn);
      m_run[r] = mn;
#pragma unroll
      for (int nt = 0; nt < 8; ++nt) o[nt][r] *= al;
      const float p0 = __expf(s0 - mn);
      const float p1 = __expf(s1 - mn);
      sacc[0][r] = p0;
      sacc[1][r] = p1;
      float ps = p0 + p1;
#pragma unroll
      for (int msk = 8; msk >= 1; msk >>= 1) ps += __shfl_xor(ps, msk, 32);
      l_run[r] = l_run[r] * al + ps;
    }

    // P (C-layout) -> wave-private LDS -> A-frag layout
#pragma unroll
    for (int nt = 0; nt < 2; ++nt)
#pragma unroll
      for (int r = 0; r < 8; ++r)
        Ps[wave][r + 8 * hs][nt * 16 + sub] = (h16)sacc[nt][r];
    asm volatile("s_wait_dscnt 0" ::: "memory");  // intra-wave LDS RAW
    v16h pf = load_frag_a(&Ps[wave][0][0], 40, 0);

    // O += P V : 8 output tiles along HD
#pragma unroll
    for (int nt = 0; nt < 8; ++nt) {
      v16h vf = load_frag_b(&Vt[buf][nt * 16][0], 40, 0);
      o[nt] = wmma_f16(pf, vf, o[nt]);
    }

    wait_async();
    __syncthreads();
    buf ^= 1;
  }

#pragma unroll
  for (int nt = 0; nt < 8; ++nt)
#pragma unroll
    for (int r = 0; r < 8; ++r) {
      const int trow = t0 + wave * 16 + r + 8 * hs;
      const int e    = nt * 16 + sub;
      Out[((size_t)b * T_ + trow) * D_ + head * HD_ + e] = o[nt][r] / l_run[r];
    }
}

// ---------------------------------------------------------------------------
// Host-side launcher
// ---------------------------------------------------------------------------
extern "C" void kernel_launch(void* const* d_in, const int* in_sizes, int n_in,
                              void* d_out, int out_size, void* d_ws, size_t ws_size,
                              hipStream_t stream) {
  const float* x   = (const float*)d_in[0];
  const float* Wq  = (const float*)d_in[1];
  const float* bq  = (const float*)d_in[2];
  const float* Wk  = (const float*)d_in[3];
  const float* bk  = (const float*)d_in[4];
  const float* Wv  = (const float*)d_in[5];
  const float* bv  = (const float*)d_in[6];
  const float* W1  = (const float*)d_in[7];
  const float* b1  = (const float*)d_in[8];
  const float* W2  = (const float*)d_in[9];
  const float* b2  = (const float*)d_in[10];
  const float* g1  = (const float*)d_in[11];
  const float* be1 = (const float*)d_in[12];
  const float* g2  = (const float*)d_in[13];
  const float* be2 = (const float*)d_in[14];
  float* out = (float*)d_out;

  char* ws = (char*)d_ws;
  size_t off = 0;
  auto alloc = [&](size_t bytes) -> char* {
    char* p = ws + off;
    off += (bytes + 255) & ~(size_t)255;
    return p;
  };
  h16*   h1   = (h16*)alloc((size_t)BT_ * D_ * 2);
  h16*   wqT  = (h16*)alloc((size_t)H_ * HD_ * D_ * 2);   // [H, HD, D]
  h16*   wkT  = (h16*)alloc((size_t)H_ * HD_ * D_ * 2);
  h16*   wvT  = (h16*)alloc((size_t)H_ * HD_ * D_ * 2);
  h16*   w1T  = (h16*)alloc((size_t)DFF_ * D_ * 2);       // [DFF, D]
  h16*   w2T  = (h16*)alloc((size_t)D_ * DFF_ * 2);       // [D, DFF]
  h16*   Qb   = (h16*)alloc((size_t)BT_ * D_ * 2);        // [B,H,T,HD]
  h16*   Kb   = (h16*)alloc((size_t)BT_ * D_ * 2);        // [B,H,T,HD]
  h16*   Vb   = (h16*)alloc((size_t)BT_ * D_ * 2);        // [B,H,HD,T]
  float* attn = (float*)alloc((size_t)BT_ * D_ * 4);
  float* xres = (float*)alloc((size_t)BT_ * D_ * 4);
  h16*   h2   = (h16*)alloc((size_t)BT_ * D_ * 2);
  h16*   a1   = (h16*)alloc((size_t)BT_ * DFF_ * 2);

  const dim3 blk(256);

  // transpose-cast all weights to f16 [N,K] layout
  k_cast_t<<<dim3(HD_ / 32, D_ / 32, H_), blk, 0, stream>>>(Wq, wqT, D_, HD_);
  k_cast_t<<<dim3(HD_ / 32, D_ / 32, H_), blk, 0, stream>>>(Wk, wkT, D_, HD_);
  k_cast_t<<<dim3(HD_ / 32, D_ / 32, H_), blk, 0, stream>>>(Wv, wvT, D_, HD_);
  k_cast_t<<<dim3(DFF_ / 32, D_ / 32, 1), blk, 0, stream>>>(W1, w1T, D_, DFF_);
  k_cast_t<<<dim3(D_ / 32, DFF_ / 32, 1), blk, 0, stream>>>(W2, w2T, DFF_, D_);

  // LN1
  k_ln<0><<<dim3(BT_), blk, 0, stream>>>(x, nullptr, g1, be1, h1, nullptr);

  // QKV projections (V written transposed)
  k_gemm_qkv<0><<<dim3(BT_ / 128, H_), blk, 0, stream>>>(h1, wqT, bq, Qb);
  k_gemm_qkv<0><<<dim3(BT_ / 128, H_), blk, 0, stream>>>(h1, wkT, bk, Kb);
  k_gemm_qkv<1><<<dim3(BT_ / 128, H_), blk, 0, stream>>>(h1, wvT, bv, Vb);

  // causal flash attention
  k_attn<<<dim3(T_ / 128, H_, B_), blk, 0, stream>>>(Qb, Kb, Vb, attn);

  // residual + LN2
  k_ln<1><<<dim3(BT_), blk, 0, stream>>>(x, attn, g2, be2, h2, xres);

  // FFN
  k_gemm<1, 1, 0><<<dim3(DFF_ / 128, BT_ / 128), blk, 0, stream>>>(
      h2, w1T, b1, nullptr, (void*)a1, BT_, DFF_, D_);
  k_gemm<0, 0, 1><<<dim3(D_ / 128, BT_ / 128), blk, 0, stream>>>(
      a1, w2T, b2, xres, (void*)out, BT_, D_, DFF_);
}
